// NeuralMessagePassing_31404800869052
// MI455X (gfx1250) — matrix operand order repaired and checked by reading, not compiled, not run on hardware
//
#include <hip/hip_runtime.h>
#include <hip/hip_bf16.h>
#include <cstdint>
#include <cstddef>

typedef __attribute__((ext_vector_type(16))) _Float16 v16h;
typedef __attribute__((ext_vector_type(8)))  float    v8f;

#define NA    5000          // atoms
#define NE    4             // edge types
#define DH    128           // hidden
#define KPAD  5024          // K extent (multiple of 32)
#define NROW  5056          // hT row stride = KPAD + 32 (prefetch overrun pad)
#define WPR   (KPAD / 32)   // 157 bit-words per adjacency row
#define TSTEPS 4

// ---------------------------------------------------------------------------
// Pack adjacency 0/1 int32 -> 1 bit per element. [E*NA][WPR] words.
// One-time pass: reads 400MB, writes 12.5MB; packed A then lives in L2.
// ---------------------------------------------------------------------------
__global__ __launch_bounds__(256) void mpnn_pack_bits(
    const int* __restrict__ adjs, uint32_t* __restrict__ Abits)
{
  int widx = blockIdx.x * blockDim.x + threadIdx.x;
  if (widx >= NE * NA * WPR) return;
  int w    = widx % WPR;
  int row  = widx / WPR;              // e*NA + v
  int ub   = w * 32;
  const int* p = adjs + (size_t)row * NA + ub;
  int nvalid = NA - ub; if (nvalid > 32) nvalid = 32;
  uint32_t bits = 0;
#pragma unroll 8
  for (int j = 0; j < nvalid; ++j) bits |= (uint32_t)(p[j] & 1) << j;
  Abits[widx] = bits;
}

// ---------------------------------------------------------------------------
// Init: embedding lookup -> h0 (f32 [N,DH]) + transposed f16 copy hT0
// ([DH,NROW]); zero padding columns of BOTH hT buffers.
// ---------------------------------------------------------------------------
__global__ __launch_bounds__(256) void mpnn_init_embed(
    const int* __restrict__ atom, const float* __restrict__ emb,
    float* __restrict__ h0, _Float16* __restrict__ hT0, _Float16* __restrict__ hT1)
{
  int idx = blockIdx.x * blockDim.x + threadIdx.x;
  if (idx >= DH * NROW) return;
  int v = idx % NROW;
  int i = idx / NROW;
  if (v < NA) {
    float val = emb[(size_t)atom[v] * DH + i];
    h0[(size_t)v * DH + i] = val;
    hT0[(size_t)i * NROW + v] = (_Float16)val;
  } else {
    hT0[(size_t)i * NROW + v] = (_Float16)0.f;
    hT1[(size_t)i * NROW + v] = (_Float16)0.f;
  }
}

__global__ __launch_bounds__(256) void mpnn_init_H(
    const float* __restrict__ He, _Float16* __restrict__ Hh)
{
  int idx = blockIdx.x * blockDim.x + threadIdx.x;
  if (idx >= NE * DH * DH) return;
  Hh[idx] = (_Float16)He[idx];
}

// ---------------------------------------------------------------------------
// One fused message-passing step.
// Block = 128 threads = 4 waves = ONE 32-row v-tile (2 M-tiles); wave w
// computes edge type e = w for stage 1.  Each B fragment (hT) is loaded once
// and consumed by 2 wmmas (one per M-tile) -> halves operand-bandwidth
// demand; identical B addresses across the 4 waves reuse WGP$.
// Per 32-K step: 2 b32 bit loads + unpack, 16 wmmas, then the 16-load clause
// for the next step (covered by the wmmas in flight).
// ---------------------------------------------------------------------------
__global__ __launch_bounds__(128, 1) void mpnn_step(
    const uint32_t* __restrict__ Abits,   // [E*NA][WPR] packed adjacency
    const float*    __restrict__ h_prev,  // [N,DH] f32
    const _Float16* __restrict__ hT,      // [DH,NROW] f16 (transposed h)
    const _Float16* __restrict__ Hh,      // [E,DH,DH] f16
    float*          __restrict__ h_out,   // [N,DH] f32
    _Float16*       __restrict__ hT_out)  // [DH,NROW] f16
{
  __shared__ float lds_agg[NE][2][16 * DH];  // 64 KB: agg per (e, m-tile)

  const int wave = threadIdx.x >> 5;      // == edge type e for stage 1
  const int lane = threadIdx.x & 31;
  const int vbase = blockIdx.x * 32;      // 32-row block tile
  const int lrow = lane & 15;             // M / N index within 16
  const int hi   = lane >> 4;             // lane half (0 or 1)
  const int g    = hi * 16;               // A-frag K-chunk shift

  const int      arow0  = vbase + lrow;        // M-tile 0 row
  const int      arow1  = arow0 + 16;          // M-tile 1 row
  const bool     ok0    = arow0 < NA;
  const bool     ok1    = arow1 < NA;
  const uint32_t msk0   = ok0 ? 0xFFFFFFFFu : 0u;
  const uint32_t msk1   = ok1 ? 0xFFFFFFFFu : 0u;
  const int      arow0c = ok0 ? arow0 : 0;
  const int      arow1c = ok1 ? arow1 : 0;

  // ---------------- Stage 1: agg_e = A_e @ h ----------------------------
  const uint32_t* Aw0 = Abits + (size_t)(wave * NA + arow0c) * WPR;
  const uint32_t* Aw1 = Abits + (size_t)(wave * NA + arow1c) * WPR;
  const _Float16* bb  = hT + hi * 16;     // per-lane-half K base
  const size_t    cs  = (size_t)NROW;     // column stride (elements)

  v8f acc[2][8] = {};
  v16h bf[8];

  // Prologue: B fragment set for ub = 0.
#pragma unroll
  for (int nt = 0; nt < 8; ++nt)
    bf[nt] = *(const v16h*)(bb + (size_t)(nt * 16 + lrow) * cs);

  for (int ub = 0; ub < KPAD; ub += 32) {
    const int w = ub >> 5;
    uint32_t bits0 = (Aw0[w] & msk0) >> g;
    uint32_t bits1 = (Aw1[w] & msk1) >> g;

    // A fragments (contiguous k = 16h + s layout, pre-shifted by g: needed
    // bits at positions {t} and {8+t}).  f16(1.0) is exact.
    v16h af0, af1;
#pragma unroll
    for (int t = 0; t < 8; ++t) {
      af0[t]     = (_Float16)(unsigned short)((bits0 >> t) & 1u);
      af0[8 + t] = (_Float16)(unsigned short)((bits0 >> (8 + t)) & 1u);
      af1[t]     = (_Float16)(unsigned short)((bits1 >> t) & 1u);
      af1[8 + t] = (_Float16)(unsigned short)((bits1 >> (8 + t)) & 1u);
    }

    // 16 wmmas: each resident B fragment feeds both M-tiles.
#pragma unroll
    for (int nt = 0; nt < 8; ++nt) {
      acc[0][nt] = __builtin_amdgcn_wmma_f32_16x16x32_f16(
          false, af0, false, bf[nt], (short)0, acc[0][nt], false, false);
      acc[1][nt] = __builtin_amdgcn_wmma_f32_16x16x32_f16(
          false, af1, false, bf[nt], (short)0, acc[1][nt], false, false);
    }

    // Load next k-step's B fragments (after the wmmas in program order so
    // the clause is covered by XDL execution).  hT rows have 32 zero pad
    // elements, so the final overrun read is in-bounds.
    const _Float16* bn = bb + (ub + 32);
#pragma unroll
    for (int nt = 0; nt < 8; ++nt)
      bf[nt] = *(const v16h*)(bn + (size_t)(nt * 16 + lrow) * cs);
  }

  // Spill agg tiles to LDS in C-layout (row M = r + hi*8, col = lrow)
#pragma unroll
  for (int mt = 0; mt < 2; ++mt) {
    float* aggw = lds_agg[wave][mt];
#pragma unroll
    for (int nt = 0; nt < 8; ++nt) {
#pragma unroll
      for (int r = 0; r < 8; ++r) {
        aggw[(r + hi * 8) * DH + nt * 16 + lrow] = acc[mt][nt][r];
      }
    }
  }
  __syncthreads();

  // ---------------- Stage 2: msg = sum_e agg_e @ H_e^T ------------------
  // Each wave produces 2 of the 8 output i-tiles for both M-tiles.
  v8f macc[2][2] = {};
  for (int e = 0; e < NE; ++e) {
    for (int jb = 0; jb < DH; jb += 32) {
      // B[k=j][n=i] = H_e[i,j]: contiguous in j -> 32B per lane; load the
      // two i-tile fragments once and reuse across both M-tiles.
      const _Float16* hb = Hh + (size_t)e * DH * DH + jb + hi * 16;
      v16h bfi[2];
#pragma unroll
      for (int m = 0; m < 2; ++m) {
        const int icol = (wave * 2 + m) * 16 + lrow;
        bfi[m] = *(const v16h*)(hb + (size_t)icol * DH);
      }
#pragma unroll
      for (int mt = 0; mt < 2; ++mt) {
        const float* q = lds_agg[e][mt] + lrow * DH + jb;
        v16h af2;
#pragma unroll
        for (int t = 0; t < 8; ++t) {
          af2[t]     = (_Float16)q[g + t];
          af2[8 + t] = (_Float16)q[g + 8 + t];
        }
#pragma unroll
        for (int m = 0; m < 2; ++m) {
          macc[mt][m] = __builtin_amdgcn_wmma_f32_16x16x32_f16(
              false, af2, false, bfi[m], (short)0, macc[mt][m], false, false);
        }
      }
    }
  }

  // ---------------- Epilogue: h' = sigmoid(h + msg) ---------------------
#pragma unroll
  for (int mt = 0; mt < 2; ++mt) {
#pragma unroll
    for (int m = 0; m < 2; ++m) {
#pragma unroll
      for (int r = 0; r < 8; ++r) {
        const int v = vbase + mt * 16 + r + hi * 8;
        const int i = (wave * 2 + m) * 16 + lrow;
        if (v < NA) {
          float x = h_prev[(size_t)v * DH + i] + macc[mt][m][r];
          float y = 1.0f / (1.0f + __expf(-x));
          h_out[(size_t)v * DH + i] = y;
          hT_out[(size_t)i * NROW + v] = (_Float16)y;
        }
      }
    }
  }
}

// ---------------------------------------------------------------------------
// Host-side launch (graph-capture safe: only async launches on `stream`)
// ---------------------------------------------------------------------------
extern "C" void kernel_launch(void* const* d_in, const int* in_sizes, int n_in,
                              void* d_out, int out_size, void* d_ws, size_t ws_size,
                              hipStream_t stream) {
  (void)in_sizes; (void)n_in; (void)out_size; (void)ws_size;

  const int*   atom_arr  = (const int*)d_in[0];
  const int*   adjs      = (const int*)d_in[1];
  /* d_in[2] = max_time_steps (device scalar); T is 4 per the reference. */
  const float* embedding = (const float*)d_in[3];
  const float* H_edges   = (const float*)d_in[4];
  float* out = (float*)d_out;

  // Workspace layout (all offsets 32B-aligned):
  //   Abits : E*NA*WPR u32  (12,560,000 B)
  //   hT0   : DH*NROW f16   ( 1,294,336 B)
  //   hT1   : DH*NROW f16   ( 1,294,336 B)
  //   Hh    : E*DH*DH f16   (   131,072 B)
  //   h0    : NA*DH  f32    ( 2,560,000 B)
  const size_t AB_BYTES = (size_t)NE * NA * WPR * sizeof(uint32_t);
  const size_t HT_BYTES = (size_t)DH * NROW * sizeof(_Float16);
  const size_t HH_BYTES = (size_t)NE * DH * DH * sizeof(_Float16);
  char* ws = (char*)d_ws;
  uint32_t* Abits = (uint32_t*)ws;
  _Float16* hT0   = (_Float16*)(ws + AB_BYTES);
  _Float16* hT1   = (_Float16*)(ws + AB_BYTES + HT_BYTES);
  _Float16* Hh    = (_Float16*)(ws + AB_BYTES + 2 * HT_BYTES);
  float*    h0    = (float*)   (ws + AB_BYTES + 2 * HT_BYTES + HH_BYTES);

  {
    int total = NE * NA * WPR;
    mpnn_pack_bits<<<(total + 255) / 256, 256, 0, stream>>>(adjs, Abits);
  }
  {
    int total = DH * NROW;
    mpnn_init_embed<<<(total + 255) / 256, 256, 0, stream>>>(
        atom_arr, embedding, h0, hT0, hT1);
  }
  {
    int total = NE * DH * DH;
    mpnn_init_H<<<(total + 255) / 256, 256, 0, stream>>>(H_edges, Hh);
  }

  const int vblocks = (NA + 31) / 32;         // 157 blocks, 4 waves each
  const float* hp = h0;
  for (int t = 0; t < TSTEPS; ++t) {
    _Float16* hT_in  = (t & 1) ? hT1 : hT0;
    _Float16* hT_nxt = (t & 1) ? hT0 : hT1;
    float* h_new = out + (size_t)t * NA * DH;
    mpnn_step<<<vblocks, 128, 0, stream>>>(
        Abits, hp, hT_in, Hh, h_new, hT_nxt);
    hp = h_new;
  }
}
